// TransformerEncoder_32031866094073
// MI455X (gfx1250) — matrix operand order repaired and checked by reading, not compiled, not run on hardware
//
#include <hip/hip_runtime.h>
#include <hip/hip_bf16.h>
#include <stdint.h>
#include <math.h>

// ---------------------------------------------------------------------------
// CDNA5 (gfx1250, wave32) transformer encoder.
// All GEMMs use v_wmma_f32_16x16x32_f16; attention is flash-style (no SxS
// score tensor). Weights are pre-transposed at f32->f16 conversion time so
// every LDS staging op in the hot loops is a pure 64B vector copy.
// ---------------------------------------------------------------------------

typedef _Float16 f16;
typedef __attribute__((ext_vector_type(16))) _Float16 v16h;
typedef __attribute__((ext_vector_type(8)))  _Float16 v8h;
typedef __attribute__((ext_vector_type(8)))  float    v8f;

#define WMMA_F16(a, b, c) \
  __builtin_amdgcn_wmma_f32_16x16x32_f16(false, (a), false, (b), (short)0, (c), false, false)

// A fragment (16x32, f16). ISA layout: lane<16 -> row=lane, k={0..7,16..23};
// lane>=16 -> row=lane-16, k={8..15,24..31}. p0 points at (row0,k0), ld in halves.
__device__ __forceinline__ v16h load_a_frag(const f16* p0, int ld, int lane) {
  int hf = lane >> 4, l = lane & 15;
  const f16* p = p0 + (size_t)l * ld + hf * 8;
  v8h lo = *(const v8h*)p;
  v8h hi = *(const v8h*)(p + 16);
  v16h r;
#pragma unroll
  for (int i = 0; i < 8; i++) { r[i] = lo[i]; r[i + 8] = hi[i]; }
  return r;
}

// B fragment (32x16, f16) from n-major storage (rows = N, contiguous in K).
// ISA layout: lane<16 -> col=lane, k={0..7,16..23}; lane>=16 -> col=lane-16, k={8..15,24..31}.
__device__ __forceinline__ v16h load_b_frag(const f16* p0, int ld, int lane) {
  int hf = lane >> 4, l = lane & 15;
  const f16* p = p0 + (size_t)l * ld + hf * 8;
  v8h lo = *(const v8h*)p;
  v8h hi = *(const v8h*)(p + 16);
  v16h r;
#pragma unroll
  for (int i = 0; i < 8; i++) { r[i] = lo[i]; r[i + 8] = hi[i]; }
  return r;
}

// ---------------------------------------------------------------------------
// Tiled WMMA GEMM: C[M,N] = A[M,K] @ Bt[N,K]^T   (A,Bt f16 row-major)
// flags: bit0 = add bias[col], bit1 = relu.  resid/Cf/Ch/Cht nullable.
// Cht: transposed f16 store for attention-V: row=(b,s), col=(h,e) ->
//      Cht[((b*H + h)*TE + e)*TS + s]   (TS = S, TE = E, H = N/TE).
// Block = 256 threads (8 waves, 2x4 wave grid), tile 128x128, k-step 64.
// ---------------------------------------------------------------------------
__launch_bounds__(256)
__global__ void gemm_f16(const f16* __restrict__ A, const f16* __restrict__ Bt,
                         const float* __restrict__ bias,
                         const float* __restrict__ resid,
                         float* __restrict__ Cf, f16* __restrict__ Ch,
                         f16* __restrict__ Cht,
                         int M, int N, int K, int TS, int TE, int flags) {
  const int LD = 72;                    // 64 + 8 pad halves; 144B rows (16B aligned)
  __shared__ f16 As[128 * 72];          // [m][k]
  __shared__ f16 Bs[128 * 72];          // [n][k]

  int tid = threadIdx.x;
  int lane = tid & 31, wid = tid >> 5;
  int wm = wid >> 2;                    // 0..1 -> 64 rows
  int wn = wid & 3;                     // 0..3 -> 32 cols
  int tile_m = blockIdx.x * 128;
  int tile_n = blockIdx.y * 128;

  int srow = tid >> 1;                  // staging row 0..127
  int sch  = (tid & 1) * 32;            // staging chunk (halves)

  v8f acc[4][2] = {};

  for (int k0 = 0; k0 < K; k0 += 64) {
    // --- stage A and Bt tiles: each thread copies one 64B chunk per array ---
    {
      const f16* ga = A + (size_t)(tile_m + srow) * K + k0 + sch;
      v8h a0 = *(const v8h*)ga,        a1 = *(const v8h*)(ga + 8);
      v8h a2 = *(const v8h*)(ga + 16), a3 = *(const v8h*)(ga + 24);
      f16* sa = &As[srow * LD + sch];
      *(v8h*)sa = a0; *(v8h*)(sa + 8) = a1;
      *(v8h*)(sa + 16) = a2; *(v8h*)(sa + 24) = a3;

      const f16* gb = Bt + (size_t)(tile_n + srow) * K + k0 + sch;
      v8h b0 = *(const v8h*)gb,        b1 = *(const v8h*)(gb + 8);
      v8h b2 = *(const v8h*)(gb + 16), b3 = *(const v8h*)(gb + 24);
      f16* sb = &Bs[srow * LD + sch];
      *(v8h*)sb = b0; *(v8h*)(sb + 8) = b1;
      *(v8h*)(sb + 16) = b2; *(v8h*)(sb + 24) = b3;
    }
    __syncthreads();

    if (k0 + 64 < K) {                  // prefetch next k-tile (global_prefetch_b8)
      __builtin_prefetch(A + (size_t)(tile_m + srow) * K + k0 + 64 + sch, 0, 0);
      __builtin_prefetch(Bt + (size_t)(tile_n + srow) * K + k0 + 64 + sch, 0, 0);
    }

#pragma unroll
    for (int ks = 0; ks < 64; ks += 32) {
      v16h bf0 = load_b_frag(&Bs[(wn * 32) * LD + ks], LD, lane);
      v16h bf1 = load_b_frag(&Bs[(wn * 32 + 16) * LD + ks], LD, lane);
      v16h af0 = load_a_frag(&As[(wm * 64) * LD + ks], LD, lane);
      v16h af1 = load_a_frag(&As[(wm * 64 + 16) * LD + ks], LD, lane);
      v16h af2 = load_a_frag(&As[(wm * 64 + 32) * LD + ks], LD, lane);
      v16h af3 = load_a_frag(&As[(wm * 64 + 48) * LD + ks], LD, lane);
      acc[0][0] = WMMA_F16(af0, bf0, acc[0][0]);
      acc[0][1] = WMMA_F16(af0, bf1, acc[0][1]);
      acc[1][0] = WMMA_F16(af1, bf0, acc[1][0]);
      acc[1][1] = WMMA_F16(af1, bf1, acc[1][1]);
      acc[2][0] = WMMA_F16(af2, bf0, acc[2][0]);
      acc[2][1] = WMMA_F16(af2, bf1, acc[2][1]);
      acc[3][0] = WMMA_F16(af3, bf0, acc[3][0]);
      acc[3][1] = WMMA_F16(af3, bf1, acc[3][1]);
    }
    __syncthreads();
  }

  // --- epilogue: C layout: VGPR e -> row e (lanes 0-15) / row 8+e (lanes 16-31)
  int hf = lane >> 4, l = lane & 15;
#pragma unroll
  for (int r = 0; r < 4; r++) {
#pragma unroll
    for (int c = 0; c < 2; c++) {
      int col = tile_n + wn * 32 + c * 16 + l;
      float bv = (flags & 1) ? bias[col] : 0.f;
#pragma unroll
      for (int e = 0; e < 8; e++) {
        int row = tile_m + wm * 64 + r * 16 + hf * 8 + e;
        float v = acc[r][c][e] + bv;
        if (flags & 2) v = fmaxf(v, 0.f);
        size_t idx = (size_t)row * N + col;
        if (resid) v += resid[idx];
        if (Cf) Cf[idx] = v;
        if (Ch) Ch[idx] = (f16)v;
        if (Cht) {                      // V^T layout [b][h][e][s]
          int s = row % TS, bi = row / TS;
          int ee = col % TE, hi = col / TE;
          size_t ti = (((size_t)bi * (N / TE) + hi) * TE + ee) * TS + s;
          Cht[ti] = (f16)v;
        }
      }
    }
  }
}

// ---------------------------------------------------------------------------
// Flash attention: softmax(Q K^T / scale + bias) V, online softmax.
// Q/K f16 [B,S,H,E] (E=128); Vt f16 [B,H,E,S] (pre-transposed by GEMM).
// Block: 8 waves, 128 q rows (16/wave); t-blocks of 64. Output f16 [B,S,H*E].
// ---------------------------------------------------------------------------
__launch_bounds__(256)
__global__ void attn_kernel(const f16* __restrict__ Q, const f16* __restrict__ Km,
                            const f16* __restrict__ Vt,
                            const uint8_t* __restrict__ mask,
                            f16* __restrict__ O,
                            int Bn, int Sn, int Hn, int En, float inv_scale) {
  const int LDK = 136;  // E + 8 pad  (272 B rows)
  const int LDV = 72;   // 64 + 8 pad (144 B rows)
  const int LDP = 72;
  __shared__ f16 Ks[64 * 136];     // [t][e] == n-major for Q K^T
  __shared__ f16 Vs[128 * 72];     // [e][t] == n-major for P V
  __shared__ f16 Ps[8 * 16 * 72];  // per-wave P scratch (C-layout -> A-layout)
  __shared__ float sbias[64];

  int tid = threadIdx.x, lane = tid & 31, wid = tid >> 5;
  int hf = lane >> 4, l = lane & 15;
  int bh = blockIdx.y;
  int b = bh / Hn, h = bh % Hn;
  int q0 = blockIdx.x * 128 + wid * 16;
  const size_t ldq = (size_t)Hn * En;

  // Q fragments for this wave's 16 rows (4 k-steps over E=128), from global.
  v16h qf[4];
  {
    const f16* qb = Q + ((size_t)(b * Sn + q0) * Hn + h) * En;
#pragma unroll
    for (int ka = 0; ka < 4; ka++) {
      const f16* p = qb + (size_t)l * ldq + ka * 32 + hf * 8;
      v8h lo = *(const v8h*)p;
      v8h hi = *(const v8h*)(p + 16);
      v16h r;
#pragma unroll
      for (int i = 0; i < 8; i++) { r[i] = lo[i]; r[i + 8] = hi[i]; }
      qf[ka] = r;
    }
  }

  v8f of[8] = {};
  float m8[8], l8[8];
#pragma unroll
  for (int e = 0; e < 8; e++) { m8[e] = -3.0e38f; l8[e] = 0.f; }

  for (int t0 = 0; t0 < Sn; t0 += 64) {
    // --- stage K rows (row-major by t) -- pure 64B copies ---
    {
      int tr = tid >> 2, ck = (tid & 3) * 32;   // 64 t-rows x 4 chunks
      const f16* gk = Km + ((size_t)(b * Sn + t0 + tr) * Hn + h) * En + ck;
      v8h k0v = *(const v8h*)gk,        k1v = *(const v8h*)(gk + 8);
      v8h k2v = *(const v8h*)(gk + 16), k3v = *(const v8h*)(gk + 24);
      f16* sp = &Ks[tr * LDK + ck];
      *(v8h*)sp = k0v; *(v8h*)(sp + 8) = k1v;
      *(v8h*)(sp + 16) = k2v; *(v8h*)(sp + 24) = k3v;

      // --- stage V^T rows ([e][t]) -- pure 64B copies ---
      int er = tid >> 1, cv = (tid & 1) * 32;   // 128 e-rows x 2 chunks
      const f16* gv = Vt + (((size_t)(b * Hn + h) * En + er) * Sn) + t0 + cv;
      v8h v0 = *(const v8h*)gv,        v1 = *(const v8h*)(gv + 8);
      v8h v2 = *(const v8h*)(gv + 16), v3 = *(const v8h*)(gv + 24);
      f16* vp = &Vs[er * LDV + cv];
      *(v8h*)vp = v0; *(v8h*)(vp + 8) = v1;
      *(v8h*)(vp + 16) = v2; *(v8h*)(vp + 24) = v3;

      if (tid < 64)
        sbias[tid] = mask[(size_t)b * Sn + t0 + tid] ? 0.f : -3.0e38f;
    }
    __syncthreads();

    // --- S = Q K^T / scale + bias  (4 t-chunks of 16 cols) ---
    v8f sf[4];
#pragma unroll
    for (int tn = 0; tn < 4; tn++) {
      v8f a = {};
#pragma unroll
      for (int ka = 0; ka < 4; ka++) {
        v16h bf = load_b_frag(&Ks[(tn * 16) * LDK + ka * 32], LDK, lane);
        a = WMMA_F16(qf[ka], bf, a);
      }
#pragma unroll
      for (int e = 0; e < 8; e++)
        a[e] = a[e] * inv_scale + sbias[tn * 16 + l];
      sf[tn] = a;
    }

    // --- online softmax: stats are per C-row = per accumulator element e ---
#pragma unroll
    for (int e = 0; e < 8; e++) {
      float rmax = m8[e];
#pragma unroll
      for (int tn = 0; tn < 4; tn++) rmax = fmaxf(rmax, sf[tn][e]);
#pragma unroll
      for (int off = 1; off < 16; off <<= 1)
        rmax = fmaxf(rmax, __shfl_xor(rmax, off, 32));
      float corr = __expf(m8[e] - rmax);
      m8[e] = rmax;
      l8[e] *= corr;
#pragma unroll
      for (int oe = 0; oe < 8; oe++) of[oe][e] *= corr;
    }
    float rsum[8];
#pragma unroll
    for (int e = 0; e < 8; e++) rsum[e] = 0.f;
#pragma unroll
    for (int tn = 0; tn < 4; tn++)
#pragma unroll
      for (int e = 0; e < 8; e++) {
        float p = __expf(sf[tn][e] - m8[e]);
        sf[tn][e] = p;
        rsum[e] += p;
      }
#pragma unroll
    for (int e = 0; e < 8; e++) {
#pragma unroll
      for (int off = 1; off < 16; off <<= 1)
        rsum[e] += __shfl_xor(rsum[e], off, 32);
      l8[e] += rsum[e];
    }

    // --- P: C-layout -> per-wave LDS -> A-layout (same-wave LDS is in-order) ---
    f16* pw = &Ps[wid * 16 * LDP];
#pragma unroll
    for (int tn = 0; tn < 4; tn++)
#pragma unroll
      for (int e = 0; e < 8; e++)
        pw[(e + hf * 8) * LDP + tn * 16 + l] = (f16)sf[tn][e];

    v16h pf[2];
#pragma unroll
    for (int kp = 0; kp < 2; kp++)
      pf[kp] = load_a_frag(pw + kp * 32, LDP, lane);

    // --- O += P @ V ---
#pragma unroll
    for (int oe = 0; oe < 8; oe++) {
#pragma unroll
      for (int kp = 0; kp < 2; kp++) {
        v16h bf = load_b_frag(&Vs[(oe * 16) * LDV + kp * 32], LDV, lane);
        of[oe] = WMMA_F16(pf[kp], bf, of[oe]);
      }
    }
    __syncthreads();
  }

  // --- normalize + store f16 [B,S,H*E] ---
#pragma unroll
  for (int e = 0; e < 8; e++) {
    float inv = 1.0f / l8[e];
    int row = q0 + e + hf * 8;
#pragma unroll
    for (int oe = 0; oe < 8; oe++) {
      int col = oe * 16 + l;
      O[((size_t)(b * Sn + row) * Hn + h) * En + col] = (f16)(of[oe][e] * inv);
    }
  }
}

// ---------------------------------------------------------------------------
// LayerNorm over D per row; writes f32 (+ optional f16 copy).
// ---------------------------------------------------------------------------
__launch_bounds__(256)
__global__ void ln_kernel(const float* __restrict__ X, const float* __restrict__ g,
                          const float* __restrict__ be, float* __restrict__ Yf,
                          f16* __restrict__ Yh, int D) {
  __shared__ float red[8];
  size_t row = blockIdx.x;
  const float* x = X + row * D;
  int tid = threadIdx.x, lane = tid & 31, wid = tid >> 5;

  float s = 0.f;
  for (int i = tid; i < D; i += 256) s += x[i];
#pragma unroll
  for (int off = 16; off; off >>= 1) s += __shfl_xor(s, off, 32);
  if (lane == 0) red[wid] = s;
  __syncthreads();
  s = 0.f;
#pragma unroll
  for (int i = 0; i < 8; i++) s += red[i];
  float mean = s / D;
  __syncthreads();

  float v = 0.f;
  for (int i = tid; i < D; i += 256) { float d = x[i] - mean; v += d * d; }
#pragma unroll
  for (int off = 16; off; off >>= 1) v += __shfl_xor(v, off, 32);
  if (lane == 0) red[wid] = v;
  __syncthreads();
  v = 0.f;
#pragma unroll
  for (int i = 0; i < 8; i++) v += red[i];
  float rstd = rsqrtf(v / D + 1e-5f);

  for (int i = tid; i < D; i += 256) {
    float y = (x[i] - mean) * rstd * g[i] + be[i];
    Yf[row * D + i] = y;
    if (Yh) Yh[row * D + i] = (f16)y;
  }
}

// ---------------------------------------------------------------------------
// x + sinusoidal positional encoding; writes f32 and f16 copies.
// ---------------------------------------------------------------------------
__launch_bounds__(256)
__global__ void posenc_kernel(const float* __restrict__ Xin, float* __restrict__ Xf,
                              f16* __restrict__ Xh, int S, int D) {
  size_t row = blockIdx.x;          // b*S + s
  int s = (int)(row % S);
  for (int i = threadIdx.x; i < D; i += 256) {
    int ii = (i >> 1) << 1;
    float ang = (float)s * powf(10000.0f, -((float)ii) / (float)D);
    float p = (i & 1) ? __cosf(ang) : __sinf(ang);
    float v = Xin[row * D + i] + p;
    Xf[row * D + i] = v;
    Xh[row * D + i] = (f16)v;
  }
}

// ---------------------------------------------------------------------------
// f32 [K,N] -> f16 transposed [N,K], LDS-tiled 32x32 (both sides coalesced).
// Grid: (K/32, N/32), block 256.
// ---------------------------------------------------------------------------
__launch_bounds__(256)
__global__ void f32_to_f16_t_kernel(const float* __restrict__ in,
                                    f16* __restrict__ out, int K, int N) {
  __shared__ float t[32][33];
  int k0 = blockIdx.x * 32, n0 = blockIdx.y * 32;
  int lx = threadIdx.x & 31, ly = threadIdx.x >> 5;   // 8 row-groups
#pragma unroll
  for (int r = ly; r < 32; r += 8)
    t[r][lx] = in[(size_t)(k0 + r) * N + n0 + lx];
  __syncthreads();
#pragma unroll
  for (int r = ly; r < 32; r += 8)
    out[(size_t)(n0 + r) * K + k0 + lx] = (f16)t[lx][r];
}

// ---------------------------------------------------------------------------
// Host orchestration (graph-capture safe: launches only, on `stream`).
// ---------------------------------------------------------------------------
extern "C" void kernel_launch(void* const* d_in, const int* in_sizes, int n_in,
                              void* d_out, int out_size, void* d_ws, size_t ws_size,
                              hipStream_t stream) {
  (void)in_sizes; (void)n_in; (void)out_size; (void)ws_size;
  const int L = 6, B = 4, S = 2048, D = 1024, H = 8, F = 2048;
  const int E = D / H;             // 128
  const int HE = H * E;            // 1024
  const size_t BS = (size_t)B * S; // 8192 rows
  const float inv_scale = 1.0f / sqrtf((float)E);

  const float*   x_in  = (const float*)d_in[0];
  const uint8_t* mask  = (const uint8_t*)d_in[1];   // jnp bool -> 1 byte/elem
  const float*   Wq    = (const float*)d_in[2];
  const float*   Wk    = (const float*)d_in[3];
  const float*   Wv    = (const float*)d_in[4];
  const float*   Wo    = (const float*)d_in[5];
  const float*   ln1_g = (const float*)d_in[6];
  const float*   ln1_b = (const float*)d_in[7];
  const float*   W1    = (const float*)d_in[8];
  const float*   b1    = (const float*)d_in[9];
  const float*   W2    = (const float*)d_in[10];
  const float*   b2    = (const float*)d_in[11];
  const float*   ln2_g = (const float*)d_in[12];
  const float*   ln2_b = (const float*)d_in[13];

  char* ws = (char*)d_ws;
  size_t off = 0;
  auto alloc = [&](size_t bytes) -> void* {
    void* p = ws + off;
    off += (bytes + 255) & ~(size_t)255;
    return p;
  };
  float* xf  = (float*)alloc(BS * D * 4);   // current activation (f32)
  f16*   xh  = (f16*)  alloc(BS * D * 2);   // current activation (f16)
  f16*   qh  = (f16*)  alloc(BS * HE * 2);
  f16*   kh  = (f16*)  alloc(BS * HE * 2);
  f16*   vth = (f16*)  alloc(BS * HE * 2);  // V^T  [B,H,E,S]
  f16*   aoh = (f16*)  alloc(BS * HE * 2);  // attention output
  float* x2  = (float*)alloc(BS * D * 4);   // residual-sum scratch
  f16*   hh  = (f16*)  alloc(BS * F * 2);   // FFN hidden
  f16*   wtq = (f16*)  alloc((size_t)D * HE * 2);   // all weights transposed [N,K]
  f16*   wtk = (f16*)  alloc((size_t)D * HE * 2);
  f16*   wtv = (f16*)  alloc((size_t)D * HE * 2);
  f16*   wto = (f16*)  alloc((size_t)HE * D * 2);
  f16*   wt1 = (f16*)  alloc((size_t)D * F * 2);
  f16*   wt2 = (f16*)  alloc((size_t)F * D * 2);

  auto cvt_t = [&](const float* src, f16* dst, int K, int N) {
    f32_to_f16_t_kernel<<<dim3(K / 32, N / 32), 256, 0, stream>>>(src, dst, K, N);
  };

  posenc_kernel<<<dim3((unsigned)BS), 256, 0, stream>>>(x_in, xf, xh, S, D);

  for (int l = 0; l < L; l++) {
    cvt_t(Wq + (size_t)l * D * HE, wtq, D, HE);
    cvt_t(Wk + (size_t)l * D * HE, wtk, D, HE);
    cvt_t(Wv + (size_t)l * D * HE, wtv, D, HE);
    cvt_t(Wo + (size_t)l * HE * D, wto, HE, D);
    cvt_t(W1 + (size_t)l * D * F,  wt1, D, F);
    cvt_t(W2 + (size_t)l * F * D,  wt2, F, D);

    dim3 gQKV((unsigned)(BS / 128), HE / 128);
    gemm_f16<<<gQKV, 256, 0, stream>>>(xh, wtq, nullptr, nullptr, nullptr, qh,
                                       nullptr, (int)BS, HE, D, 1, 1, 0);
    gemm_f16<<<gQKV, 256, 0, stream>>>(xh, wtk, nullptr, nullptr, nullptr, kh,
                                       nullptr, (int)BS, HE, D, 1, 1, 0);
    // V projection stored directly transposed: [B,H,E,S]
    gemm_f16<<<gQKV, 256, 0, stream>>>(xh, wtv, nullptr, nullptr, nullptr, nullptr,
                                       vth, (int)BS, HE, D, S, E, 0);

    attn_kernel<<<dim3(S / 128, B * H), 256, 0, stream>>>(qh, kh, vth, mask, aoh,
                                                          B, S, H, E, inv_scale);

    // x2 = x + attn_out @ Wo
    gemm_f16<<<dim3((unsigned)(BS / 128), D / 128), 256, 0, stream>>>(
        aoh, wto, nullptr, xf, x2, nullptr, nullptr, (int)BS, D, HE, 1, 1, 0);
    // x = LN1(x2)  (f32 into xf, f16 into xh)
    ln_kernel<<<dim3((unsigned)BS), 256, 0, stream>>>(x2, ln1_g + (size_t)l * D,
                                                      ln1_b + (size_t)l * D, xf, xh, D);
    // h = relu(x @ W1 + b1)
    gemm_f16<<<dim3((unsigned)(BS / 128), F / 128), 256, 0, stream>>>(
        xh, wt1, b1 + (size_t)l * F, nullptr, nullptr, hh, nullptr,
        (int)BS, F, D, 1, 1, 3);
    // x2 = x + h @ W2 + b2
    gemm_f16<<<dim3((unsigned)(BS / 128), D / 128), 256, 0, stream>>>(
        hh, wt2, b2 + (size_t)l * D, xf, x2, nullptr, nullptr,
        (int)BS, D, F, 1, 1, 1);
    // x = LN2(x2); final layer writes f32 straight to d_out
    float* outp = (l == L - 1) ? (float*)d_out : xf;
    ln_kernel<<<dim3((unsigned)BS), 256, 0, stream>>>(x2, ln2_g + (size_t)l * D,
                                                      ln2_b + (size_t)l * D, outp, xh, D);
  }
}
